// gru_d_53523882443298
// MI455X (gfx1250) — matrix operand order repaired and checked by reading, not compiled, not run on hardware
//
#include <hip/hip_runtime.h>
#include <hip/hip_bf16.h>
#include <math.h>

// ---------------- problem constants ----------------
#define BB   1024
#define TT   128
#define VV   89
#define VP   96        // padded V (K for gamma_h GEMM)
#define HH   256
#define H3   768
#define KX   192       // padded 2V (K for gi GEMM)

typedef __bf16 bf16;
typedef bf16  v8bf  __attribute__((ext_vector_type(8)));
typedef bf16  v16bf __attribute__((ext_vector_type(16)));
typedef float v8f   __attribute__((ext_vector_type(8)));

// ---------------- fast math (native v_exp_f32 / v_rcp_f32) ----------------
__device__ __forceinline__ float fast_rcp(float x) { return __builtin_amdgcn_rcpf(x); }
__device__ __forceinline__ float fast_sigmoid(float x) {
  return fast_rcp(1.0f + __expf(-x));
}
__device__ __forceinline__ float fast_tanh(float x) {
  return 1.0f - 2.0f * fast_rcp(__expf(2.0f * x) + 1.0f);
}

// ---------------- WMMA helpers ----------------
__device__ __forceinline__ v8f wmma_bf16(v16bf a, v16bf b, v8f c) {
  return __builtin_amdgcn_wmma_f32_16x16x32_bf16(
      /*neg_a=*/false, a, /*neg_b=*/false, b,
      /*c_mod=*/(short)0, c, /*reuse_a=*/false, /*reuse_b=*/false);
}

// A fragment (16-bit A, 16x32): lane holds row m=lane&15; K elements are
// {base..base+15} with base = (lane>=16)?16:0.
__device__ __forceinline__ v16bf loadA(const bf16* rowbase, int koff) {
  v8bf lo = *(const v8bf*)(rowbase + koff);
  v8bf hi = *(const v8bf*)(rowbase + koff + 8);
  v16bf a;
#pragma unroll
  for (int i = 0; i < 8; ++i) { a[i] = lo[i]; a[i + 8] = hi[i]; }
  return a;
}

// B fragment: pre-swizzled tiles, 512 bf16 per (kt,nt) tile, lane-major:
// lane L holds 16 contiguous K (lower/upper half by L>=16) for column N=nt*16+(L&15).
__device__ __forceinline__ v16bf loadB(const bf16* base, int kt, int nt, int KT, int lane) {
  return *(const v16bf*)(base + (((size_t)(nt * KT + kt)) << 9) + lane * 16);
}

// opaque zero: makes addresses loop-variant (defeats LICM of the weight loads)
// WITHOUT destroying pointer provenance, so loads stay global_load (not flat).
__device__ __forceinline__ size_t opaque_zero() {
  unsigned long long z = 0;
  asm volatile("" : "+s"(z));
  return (size_t)z;
}

// ---------------- Phase W: weight swizzle (bf16 B-tiles) + zero loss accum ----------------
__global__ __launch_bounds__(256) void grud_prep(
    const float* __restrict__ W_h, const float* __restrict__ W_ih,
    const float* __restrict__ W_hh, float* __restrict__ numden,
    bf16* __restrict__ Bh, bf16* __restrict__ Bih, bf16* __restrict__ Bhh) {
  const int NBH  = 3 * 16 * 512;   // 96x256
  const int NBI  = 6 * 48 * 512;   // 192x768
  const int NBHH = 8 * 48 * 512;   // 256x768
  int gid = blockIdx.x * blockDim.x + threadIdx.x;
  if (gid < 256) numden[gid] = 0.0f;

  if (gid < NBH) {
    int e = gid & 15, L = (gid >> 4) & 31, tile = gid >> 9;
    int kt = tile % 3, nt = tile / 3;
    int K = kt * 32 + ((L >> 4) << 4) + e;
    int N = nt * 16 + (L & 15);
    float v = (K < VV) ? W_h[N * VV + K] : 0.0f;   // B = W_h^T
    Bh[gid] = (bf16)v;
  } else if (gid < NBH + NBI) {
    int g = gid - NBH;
    int e = g & 15, L = (g >> 4) & 31, tile = g >> 9;
    int kt = tile % 6, nt = tile / 6;
    int K = kt * 32 + ((L >> 4) << 4) + e;
    int N = nt * 16 + (L & 15);
    int ksrc = (K < 96) ? ((K < VV) ? K : -1)
                        : (((K - 96) < VV) ? (K - 7) : -1); // [x_h(0..88) | m(96..184)]
    float v = (ksrc >= 0) ? W_ih[N * (2 * VV) + ksrc] : 0.0f;
    Bih[g] = (bf16)v;
  } else if (gid < NBH + NBI + NBHH) {
    int g = gid - NBH - NBI;
    int e = g & 15, L = (g >> 4) & 31, tile = g >> 9;
    int kt = tile % 8, nt = tile / 8;
    int K = kt * 32 + ((L >> 4) << 4) + e;
    int N = nt * 16 + (L & 15);
    Bhh[g] = (bf16)W_hh[N * HH + K];
  }
}

// ---------------- Phase A: imputation scan (parallel over b,v) ----------------
__global__ __launch_bounds__(192) void grud_impute(
    const float* __restrict__ x, const float* __restrict__ mask,
    const float* __restrict__ deltas, const float* __restrict__ meanset,
    const float* __restrict__ W_x, const float* __restrict__ b_x,
    float* __restrict__ ximp, float* __restrict__ numden,
    bf16* __restrict__ Dpad, bf16* __restrict__ Xcat) {
  __shared__ float s_num, s_den;
  const int vp = threadIdx.x;                 // 0..95
  const int b  = blockIdx.x * 2 + threadIdx.y;
  const bool valid = (vp < VV);
  float wxd = 0.f, bx = 0.f, mean = 0.f;
  if (valid) { wxd = W_x[vp * VV + vp]; bx = b_x[vp]; mean = meanset[vp]; }
  float last = 0.0f;
  for (int t = 0; t < TT; ++t) {
    if (threadIdx.x == 0 && threadIdx.y == 0) { s_num = 0.0f; s_den = 0.0f; }
    __syncthreads();
    float xh = 0.f, mm = 0.f, dd = 0.f;
    if (valid) {
      size_t idx = ((size_t)b * TT + t) * VV + vp;
      float xv = x[idx], mv = mask[idx], dv = deltas[idx];
      float gx = expf(-fmaxf(dv * wxd + bx, 0.0f));
      last = (mv > 0.0f) ? xv : last;
      float xu = gx * last + (1.0f - gx) * mean;
      xh = mv * xv + (1.0f - mv) * xu;
      ximp[idx] = xh;
      mm = mv; dd = dv;
      atomicAdd(&s_num, fabsf(xv - xu) * mv);
      atomicAdd(&s_den, mv);
    }
    size_t xb = ((size_t)t * BB + b) * KX;
    Xcat[xb + vp]      = (bf16)xh;
    Xcat[xb + 96 + vp] = (bf16)mm;
    Dpad[((size_t)t * BB + b) * VP + vp] = (bf16)dd;
    __syncthreads();
    if (threadIdx.x == 0 && threadIdx.y == 0) {
      atomicAdd(&numden[t], s_num);
      atomicAdd(&numden[128 + t], s_den);
    }
  }
}

// ---------------- loss finalize ----------------
__global__ void grud_loss(const float* __restrict__ numden, float* __restrict__ out_loss) {
  __shared__ float s[128];
  int t = threadIdx.x;
  s[t] = numden[t] / (numden[128 + t] + 1e-5f);
  __syncthreads();
  for (int st = 64; st > 0; st >>= 1) { if (t < st) s[t] += s[t + st]; __syncthreads(); }
  if (t == 0) *out_loss = s[0];
}

// ---------------- Phase C: recurrent GRU, WMMA bf16 ----------------
// 64 blocks x 256 threads (8 waves). Block = 16 batch rows; wave w owns H-cols [32w,32w+32).
__global__ __launch_bounds__(256, 1) void grud_recur(
    const float* __restrict__ bias_h, const float* __restrict__ bias_ih,
    const float* __restrict__ bias_hh, const float* __restrict__ Wcls,
    const float* __restrict__ bcls,
    const bf16* __restrict__ Bh_g, const bf16* __restrict__ Bih_g,
    const bf16* __restrict__ Bhh_g, const bf16* __restrict__ Dpad,
    const bf16* __restrict__ Xcat,
    float* __restrict__ out_y, float* __restrict__ out_s) {
  __shared__ bf16  hbf[16 * HH];   // decayed h, bf16, A-source for gh
  __shared__ float hf[16 * HH];    // final h fp32 for the classification head

  const int tid  = threadIdx.x;
  const int lane = tid & 31;
  const int w    = tid >> 5;            // wave 0..7
  const int b0   = blockIdx.x * 16;
  const int c    = lane & 15;           // C-layout column within tile
  const int hi   = (lane >> 4) & 1;     // C-layout m-half
  const int m    = lane & 15;           // A-layout row
  const int kbaseA = hi ? 16 : 0;       // A-layout K base for this lane half

  // biases (constant over t) per owned tile
  float bG[2], bR[2], bZ[2], bIN[2], bHN[2];
#pragma unroll
  for (int ti = 0; ti < 2; ++ti) {
    int col = (2 * w + ti) * 16 + c;
    bG[ti]  = bias_h[col];
    bR[ti]  = bias_ih[col] + bias_hh[col];
    bZ[ti]  = bias_ih[256 + col] + bias_hh[256 + col];
    bIN[ti] = bias_ih[512 + col];
    bHN[ti] = bias_hh[512 + col];
  }

  v8f hA[2];
#pragma unroll
  for (int ti = 0; ti < 2; ++ti)
#pragma unroll
    for (int i = 0; i < 8; ++i) hA[ti][i] = 0.0f;

#pragma unroll 1
  for (int t = 0; t < TT; ++t) {
    // Per-iteration opaque zero offset: keeps the 90 B-tile loads inside the
    // loop (L2-resident global_load_b128) instead of LICM-hoisting into spills,
    // while preserving global-pointer provenance (no flat_load).
    size_t z = opaque_zero();
    const bf16* Bh  = Bh_g  + z;
    const bf16* Bih = Bih_g + z;
    const bf16* Bhh = Bhh_g + z;

    const bf16* drow = Dpad + ((size_t)t * BB + b0 + m) * VP;
    const bf16* xrow = Xcat + ((size_t)t * BB + b0 + m) * KX;
    if (t + 1 < TT) {  // prefetch next step's A rows into cache
      __builtin_prefetch(drow + (size_t)BB * VP, 0, 0);
      __builtin_prefetch(xrow + (size_t)BB * KX, 0, 0);
    }

    // ---- gamma_h = exp(-relu(d @ W_h^T + b_h)); decay h in registers ----
#pragma unroll
    for (int ti = 0; ti < 2; ++ti) {
      v8f acc;
#pragma unroll
      for (int i = 0; i < 8; ++i) acc[i] = bG[ti];
#pragma unroll
      for (int kt = 0; kt < 3; ++kt)
        acc = wmma_bf16(loadA(drow, kt * 32 + kbaseA),
                        loadB(Bh, kt, 2 * w + ti, 3, lane), acc);
#pragma unroll
      for (int i = 0; i < 8; ++i)
        hA[ti][i] *= __expf(-fmaxf(acc[i], 0.0f));
    }

    // ---- publish decayed h (bf16) for the gh GEMM A-operand ----
    __syncthreads();   // previous step's hbf readers are done
#pragma unroll
    for (int ti = 0; ti < 2; ++ti) {
      int col = (2 * w + ti) * 16 + c;
#pragma unroll
      for (int r = 0; r < 8; ++r)
        hbf[(r + 8 * hi) * HH + col] = (bf16)hA[ti][r];
    }
    __syncthreads();

    // ---- gate pre-activations: accR/accZ fuse gi+gh; n-gate parts kept apart ----
    v8f aR[2], aZ[2], aIN[2], aHN[2];
#pragma unroll
    for (int ti = 0; ti < 2; ++ti)
#pragma unroll
      for (int i = 0; i < 8; ++i) {
        aR[ti][i] = bR[ti]; aZ[ti][i] = bZ[ti];
        aIN[ti][i] = bIN[ti]; aHN[ti][i] = bHN[ti];
      }

    // gi : [16,192] x [192, 3H]
#pragma unroll
    for (int kt = 0; kt < 6; ++kt) {
      v16bf a = loadA(xrow, kt * 32 + kbaseA);
#pragma unroll
      for (int ti = 0; ti < 2; ++ti) {
        int ct = 2 * w + ti;
        aR[ti]  = wmma_bf16(a, loadB(Bih, kt, ct,      6, lane), aR[ti]);
        aZ[ti]  = wmma_bf16(a, loadB(Bih, kt, 16 + ct, 6, lane), aZ[ti]);
        aIN[ti] = wmma_bf16(a, loadB(Bih, kt, 32 + ct, 6, lane), aIN[ti]);
      }
    }
    // gh : [16,256] x [256, 3H], A from LDS
#pragma unroll
    for (int kt = 0; kt < 8; ++kt) {
      v16bf a = loadA(hbf + m * HH, kt * 32 + kbaseA);
#pragma unroll
      for (int ti = 0; ti < 2; ++ti) {
        int ct = 2 * w + ti;
        aR[ti]  = wmma_bf16(a, loadB(Bhh, kt, ct,      8, lane), aR[ti]);
        aZ[ti]  = wmma_bf16(a, loadB(Bhh, kt, 16 + ct, 8, lane), aZ[ti]);
        aHN[ti] = wmma_bf16(a, loadB(Bhh, kt, 32 + ct, 8, lane), aHN[ti]);
      }
    }

    // ---- gates + h update (elementwise, register-resident) ----
#pragma unroll
    for (int ti = 0; ti < 2; ++ti)
#pragma unroll
      for (int i = 0; i < 8; ++i) {
        float r = fast_sigmoid(aR[ti][i]);
        float z = fast_sigmoid(aZ[ti][i]);
        float n = fast_tanh(aIN[ti][i] + r * aHN[ti][i]);
        hA[ti][i] = (1.0f - z) * n + z * hA[ti][i];
      }
  }

  // ---- classification head: y = h @ W_cls^T + b ----
#pragma unroll
  for (int ti = 0; ti < 2; ++ti) {
    int col = (2 * w + ti) * 16 + c;
#pragma unroll
    for (int r = 0; r < 8; ++r)
      hf[(r + 8 * hi) * HH + col] = hA[ti][r];
  }
  __syncthreads();
  if (tid < 16) {
    float y = bcls[0];
    for (int k = 0; k < HH; ++k) y += hf[tid * HH + k] * Wcls[k];
    out_y[b0 + tid] = y;
    out_s[b0 + tid] = 1.0f / (1.0f + expf(-y));
  }
}

// ---------------- host launcher ----------------
extern "C" void kernel_launch(void* const* d_in, const int* in_sizes, int n_in,
                              void* d_out, int out_size, void* d_ws, size_t ws_size,
                              hipStream_t stream) {
  const float* x       = (const float*)d_in[0];
  const float* mask    = (const float*)d_in[1];
  const float* deltas  = (const float*)d_in[2];
  const float* meanset = (const float*)d_in[3];
  const float* W_h     = (const float*)d_in[4];
  const float* b_h     = (const float*)d_in[5];
  const float* W_x     = (const float*)d_in[6];
  const float* b_x     = (const float*)d_in[7];
  const float* W_ih    = (const float*)d_in[8];
  const float* b_ih    = (const float*)d_in[9];
  const float* W_hh    = (const float*)d_in[10];
  const float* b_hh    = (const float*)d_in[11];
  const float* W_cls   = (const float*)d_in[12];
  const float* b_cls   = (const float*)d_in[13];

  float* out = (float*)d_out;
  const size_t XIMP = (size_t)BB * TT * VV;       // 11,665,408
  float* out_loss = out + XIMP;
  float* out_y    = out + XIMP + 1;
  float* out_s    = out + XIMP + 1 + BB;

  // workspace carve (all vector-load alignments are multiples of 32B)
  char* ws = (char*)d_ws;
  float* numden = (float*)ws;                         size_t off = 1024;
  bf16* Bh   = (bf16*)(ws + off); off += (size_t)3 * 16 * 512 * 2;
  bf16* Bih  = (bf16*)(ws + off); off += (size_t)6 * 48 * 512 * 2;
  bf16* Bhh  = (bf16*)(ws + off); off += (size_t)8 * 48 * 512 * 2;
  bf16* Dpad = (bf16*)(ws + off); off += (size_t)TT * BB * VP * 2;
  bf16* Xcat = (bf16*)(ws + off); off += (size_t)TT * BB * KX * 2;

  grud_prep<<<1440, 256, 0, stream>>>(W_h, W_ih, W_hh, numden, Bh, Bih, Bhh);

  dim3 ablk(96, 2);
  grud_impute<<<BB / 2, ablk, 0, stream>>>(x, mask, deltas, meanset, W_x, b_x,
                                           out, numden, Dpad, Xcat);

  grud_loss<<<1, 128, 0, stream>>>(numden, out_loss);

  grud_recur<<<BB / 16, 256, 0, stream>>>(b_h, b_ih, b_hh, W_cls, b_cls,
                                          Bh, Bih, Bhh, Dpad, Xcat, out_y, out_s);
}
